// IAF_34660386078871
// MI455X (gfx1250) — matrix operand order repaired and checked by reading, compiled, not run on hardware
//
#include <hip/hip_runtime.h>
#include <math.h>
#include <stdint.h>

// ---------------------------------------------------------------------------
// Types for CDNA5 WMMA (wave32): v_wmma_f32_16x16x32_bf16
// ---------------------------------------------------------------------------
typedef __attribute__((ext_vector_type(16))) __bf16   v16bf;
typedef __attribute__((ext_vector_type(8)))  float    v8f;
typedef __attribute__((ext_vector_type(4)))  unsigned int u32x4;

union BF16Frag {
    v16bf v;
    u32x4 q[2];
};

#define GF_ROWSCALE 1
#define GF_BIAS     2
#define GF_ELU      4

__device__ __forceinline__ unsigned short f2bf(float f) {
    uint32_t u = __float_as_uint(f);
    uint32_t r = u + 0x7FFFu + ((u >> 16) & 1u);   // round-to-nearest-even
    return (unsigned short)(r >> 16);
}

__device__ __forceinline__ float eluf(float x) { return x > 0.f ? x : expm1f(x); }
__device__ __forceinline__ float sigm(float x) { return 1.f / (1.f + expf(-x)); }

__device__ __forceinline__ float wave_reduce(float v) {
    #pragma unroll
    for (int off = 16; off > 0; off >>= 1) v += __shfl_xor(v, off, 32);
    return v;
}

// ---------------------------------------------------------------------------
// Generic batched bf16 WMMA GEMM:  C[b] = epi( A[b] (MxK, row-major bf16)
//                                         @ Bt[b]^T (Bt is NxK row-major bf16) )
// Block = 256 threads = 8 waves arranged 2(M) x 4(N); each wave: 32 x 32 tile
// (2 M-frags x 2 N-frags, 4 accumulators -> 4 WMMAs per 8 b128 loads/lane).
// Block tile = 64 x 128.  Grid: x = N/128, y = M/64, z = batch.
// Epilogue: optional per-row scale (graph row-normalization), bias, ELU.
// ---------------------------------------------------------------------------
__global__ __launch_bounds__(256)
void k_gemm_bf16(const unsigned short* __restrict__ A,  long long sA, int lda,
                 const unsigned short* __restrict__ Bt, long long sB, int ldb,
                 float* __restrict__ C, long long sC, int ldc,
                 int M, int N, int K,
                 const float* __restrict__ rowscale, int rsStride,
                 const float* __restrict__ bias, int flags)
{
    const int batch = blockIdx.z;
    A  += (size_t)batch * sA;
    Bt += (size_t)batch * sB;
    C  += (size_t)batch * sC;
    const float* rs = (flags & GF_ROWSCALE) ? rowscale + (size_t)batch * rsStride : nullptr;

    const int wave = threadIdx.x >> 5;
    const int lane = threadIdx.x & 31;
    const int wm = wave >> 2;                 // 0..1
    const int wn = wave & 3;                  // 0..3
    const int row0 = blockIdx.y * 64 + wm * 32;
    const int col0 = blockIdx.x * 128 + wn * 32;
    const int m = lane & 15;                  // A row within frag / B,C col within frag
    const int h = lane >> 4;                  // K-half selector per ISA layouts

    v8f acc00 = {0.f,0.f,0.f,0.f,0.f,0.f,0.f,0.f};
    v8f acc01 = acc00, acc10 = acc00, acc11 = acc00;

    const unsigned short* aRow0 = A  + (size_t)(row0 + m) * lda;
    const unsigned short* aRow1 = A  + (size_t)(row0 + 16 + m) * lda;
    const unsigned short* bCol0 = Bt + (size_t)(col0 + m) * ldb;
    const unsigned short* bCol1 = Bt + (size_t)(col0 + 16 + m) * ldb;

    for (int k0 = 0; k0 < K; k0 += 32) {
        BF16Frag a0, a1, b0, b1;
        // A 16x32 bf16 layout: lane holds K = 8h..8h+7 (e0..7), 16+8h..23+8h (e8..15)
        a0.q[0] = *(const u32x4*)(aRow0 + k0 + 8 * h);
        a0.q[1] = *(const u32x4*)(aRow0 + k0 + 16 + 8 * h);
        a1.q[0] = *(const u32x4*)(aRow1 + k0 + 8 * h);
        a1.q[1] = *(const u32x4*)(aRow1 + k0 + 16 + 8 * h);
        // B 32x16 bf16 layout: lane col = m, K = 8h..8h+7 (e0..7), 16+8h..23+8h (e8..15)  (Bt is col-major of B)
        const unsigned short* bp0 = bCol0 + k0 + 8 * h;
        b0.q[0] = *(const u32x4*)(bp0);
        b0.q[1] = *(const u32x4*)(bp0 + 16);
        const unsigned short* bp1 = bCol1 + k0 + 8 * h;
        b1.q[0] = *(const u32x4*)(bp1);
        b1.q[1] = *(const u32x4*)(bp1 + 16);

        acc00 = __builtin_amdgcn_wmma_f32_16x16x32_bf16(false, a0.v, false, b0.v,
                                                        (short)0, acc00, false, false);
        acc01 = __builtin_amdgcn_wmma_f32_16x16x32_bf16(false, a0.v, false, b1.v,
                                                        (short)0, acc01, false, false);
        acc10 = __builtin_amdgcn_wmma_f32_16x16x32_bf16(false, a1.v, false, b0.v,
                                                        (short)0, acc10, false, false);
        acc11 = __builtin_amdgcn_wmma_f32_16x16x32_bf16(false, a1.v, false, b1.v,
                                                        (short)0, acc11, false, false);
    }

    const int c0 = col0 + m;
    const int c1 = col0 + 16 + m;
    const float bia0 = (flags & GF_BIAS) ? bias[c0] : 0.f;
    const float bia1 = (flags & GF_BIAS) ? bias[c1] : 0.f;

    #pragma unroll
    for (int mi = 0; mi < 2; ++mi) {
        v8f vA = mi ? acc10 : acc00;
        v8f vB = mi ? acc11 : acc01;
        const int rbase = row0 + 16 * mi + 8 * h;   // D layout: VGPR j -> row j (+8 for h=1)
        #pragma unroll
        for (int j = 0; j < 8; ++j) {
            int r = rbase + j;
            float v0 = vA[j], v1 = vB[j];
            if (rs) { float s = rs[r]; v0 *= s; v1 *= s; }
            v0 += bia0; v1 += bia1;
            if (flags & GF_ELU) { v0 = eluf(v0); v1 = eluf(v1); }
            C[(size_t)r * ldc + c0] = v0;
            C[(size_t)r * ldc + c1] = v1;
        }
    }
}

// ---------------------------------------------------------------------------
// Elementwise / helper kernels
// ---------------------------------------------------------------------------
__global__ void k_f32_to_bf16(const float* __restrict__ in,
                              unsigned short* __restrict__ out, int n) {
    int i = blockIdx.x * blockDim.x + threadIdx.x;
    if (i < n) out[i] = f2bf(in[i]);
}

// in: [batch][R][C] f32 row-major  ->  out: [batch][C][R] bf16 (col-major of in)
__global__ __launch_bounds__(256)
void k_transpose_bf16(const float* __restrict__ in, unsigned short* __restrict__ out,
                      int R, int C) {
    __shared__ float tile[32][33];
    int b = blockIdx.z;
    const float* src = in + (size_t)b * R * C;
    unsigned short* dst = out + (size_t)b * R * C;
    int c0 = blockIdx.x * 32, r0 = blockIdx.y * 32;
    int tx = threadIdx.x, ty = threadIdx.y;   // 32 x 8
    #pragma unroll
    for (int i = 0; i < 4; ++i)
        tile[ty + 8 * i][tx] = src[(size_t)(r0 + ty + 8 * i) * C + c0 + tx];
    __syncthreads();
    #pragma unroll
    for (int i = 0; i < 4; ++i)
        dst[(size_t)(c0 + ty + 8 * i) * R + r0 + tx] = f2bf(tile[tx][ty + 8 * i]);
}

// inv row-sum of adjacency (one wave per 512-wide row)
__global__ __launch_bounds__(256)
void k_rowsum_inv(const float* __restrict__ adj, float* __restrict__ inv) {
    int wave = threadIdx.x >> 5, lane = threadIdx.x & 31;
    int row = blockIdx.x * 8 + wave;
    const float* p = adj + (size_t)row * 512;
    float s = 0.f;
    for (int j = lane; j < 512; j += 32) s += p[j];
    s = wave_reduce(s);
    if (lane == 0) inv[row] = 1.f / s;
}

// Wc[t][n][k] = k<256 ? Wih[t][n][k] : Whh[t][n][k-256]   (bf16, [4][1024][512])
__global__ void k_pack_wc(const float* __restrict__ Wih, const float* __restrict__ Whh,
                          unsigned short* __restrict__ Wc, int tot) {
    int i = blockIdx.x * blockDim.x + threadIdx.x;
    if (i >= tot) return;
    int k = i & 511;
    int n = (i >> 9) & 1023;
    int t = i >> 19;
    float v = (k < 256) ? Wih[((size_t)t * 1024 + n) * 256 + k]
                        : Whh[((size_t)t * 1024 + n) * 256 + (k - 256)];
    Wc[i] = f2bf(v);
}

// z0 = eps*exp(0.5*lv)+mu ; kl = -0.5*(lv+eps^2) ; seed zc = [z | z0] in bf16
__global__ void k_z0_init(const float* __restrict__ mu, float* __restrict__ lvkl,
                          const float* __restrict__ eps, float* __restrict__ Z,
                          unsigned short* __restrict__ zc, int tot) {
    int i = blockIdx.x * blockDim.x + threadIdx.x;
    if (i >= tot) return;
    float lv = lvkl[i], m = mu[i], e = eps[i];
    float z0 = e * expf(0.5f * lv) + m;
    lvkl[i] = -0.5f * (lv + e * e);
    Z[i] = z0;
    int l = i & 255;
    size_t row = (size_t)(i >> 8);
    unsigned short b = f2bf(z0);
    zc[row * 512 + l] = b;          // z part (updated each flow step)
    zc[row * 512 + 256 + l] = b;    // z0 part (fixed)
}

// One IAF/LSTM-gate step: consumes G=[32768,1024], updates Z, KL, zc(z half)
__global__ void k_flow(const float* __restrict__ G, const float* __restrict__ bih,
                       const float* __restrict__ bhh, float* __restrict__ Z,
                       float* __restrict__ KL, unsigned short* __restrict__ zc, int tot) {
    int i = blockIdx.x * blockDim.x + threadIdx.x;
    if (i >= tot) return;
    int l = i & 255;
    size_t row = (size_t)(i >> 8);
    const float* g = G + row * 1024;
    float gi = g[l]       + bih[l]       + bhh[l];
    float gf = g[256 + l] + bih[256 + l] + bhh[256 + l];
    float gg = g[512 + l] + bih[512 + l] + bhh[512 + l];
    float go = g[768 + l] + bih[768 + l] + bhh[768 + l];
    float c  = sigm(gf) + sigm(gi) * tanhf(gg);   // c0 == 1
    float mm = sigm(go) * tanhf(c);
    float s  = sigm(c + 2.0f);
    float z  = Z[i];
    float zn = s * z + (1.f - s) * mm;
    Z[i] = zn;
    KL[i] -= logf(s + 1e-5f);
    zc[row * 512 + l] = f2bf(zn);
}

// zk[row] = sigmoid(beta * elu(dot(t2_row, Wvec) + bvec))    (one wave per row)
__global__ __launch_bounds__(256)
void k_zv(const float* __restrict__ T2, const float* __restrict__ Wvec,
          const float* __restrict__ bvec, const float* __restrict__ beta,
          float* __restrict__ ZK, int rows) {
    int wave = threadIdx.x >> 5, lane = threadIdx.x & 31;
    int row = blockIdx.x * 8 + wave;
    if (row >= rows) return;
    const float* p = T2 + (size_t)row * 256;
    float s = 0.f;
    for (int j = lane; j < 256; j += 32) s += p[j] * Wvec[j];
    s = wave_reduce(s);
    if (lane == 0) ZK[row] = sigm(beta[0] * eluf(s + bvec[0]));
}

// gz[row] = invrs[row] * sum_j adj[row][j] * zk[b][j]        (one wave per row)
__global__ __launch_bounds__(256)
void k_gz(const float* __restrict__ adj, const float* __restrict__ ZK,
          const float* __restrict__ invrs, float* __restrict__ GZ) {
    int wave = threadIdx.x >> 5, lane = threadIdx.x & 31;
    int row = blockIdx.x * 8 + wave;
    int b = row >> 9;
    const float* ap = adj + (size_t)row * 512;
    const float* zk = ZK + (size_t)b * 512;
    float s = 0.f;
    for (int j = lane; j < 512; j += 32) s += ap[j] * zk[j];
    s = wave_reduce(s);
    if (lane == 0) GZ[row] = s * invrs[row];
}

// hfin[b][l] = sum_n r[b][n][l] * zk[b][n]
__global__ __launch_bounds__(256)
void k_hfin(const float* __restrict__ R, const float* __restrict__ ZK,
            float* __restrict__ HF) {
    int b = blockIdx.x, l = threadIdx.x;
    const float* r = R + (size_t)b * 512 * 256;
    const float* zk = ZK + (size_t)b * 512;
    float acc = 0.f;
    for (int n = 0; n < 512; ++n) acc += r[(size_t)n * 256 + l] * zk[n];
    HF[b * 256 + l] = acc;
}

// per-graph MLP head + log-softmax; outputs NLL[b], ACC[b]
__global__ __launch_bounds__(256)
void k_head(const float* __restrict__ HF, const float* __restrict__ W1,
            const float* __restrict__ b1, const float* __restrict__ W2,
            const float* __restrict__ b2, const int* __restrict__ labels,
            float* __restrict__ NLL, float* __restrict__ ACC) {
    __shared__ float hf[256];
    __shared__ float a1[512];
    __shared__ float lg[10];
    int b = blockIdx.x, t = threadIdx.x;
    hf[t] = HF[b * 256 + t];
    __syncthreads();
    for (int o = t; o < 512; o += 256) {
        float s = b1[o];
        for (int l = 0; l < 256; ++l) s += hf[l] * W1[l * 512 + o];
        a1[o] = eluf(s);
    }
    __syncthreads();
    if (t < 10) {
        float s = b2[t];
        for (int o = 0; o < 512; ++o) s += a1[o] * W2[o * 10 + t];
        lg[t] = s;
    }
    __syncthreads();
    if (t == 0) {
        float mx = lg[0]; int am = 0;
        for (int j = 1; j < 10; ++j) if (lg[j] > mx) { mx = lg[j]; am = j; }
        float se = 0.f;
        for (int j = 0; j < 10; ++j) se += expf(lg[j] - mx);
        float lse = mx + logf(se);
        int lab = labels[b];
        NLL[b] = -(lg[lab] - lse);
        ACC[b] = (am == lab) ? 1.f : 0.f;
    }
}

// partial sums of mse ((r-d)^2, d recomputed from gz) and of final kl
__global__ __launch_bounds__(256)
void k_mse_kl(const float* __restrict__ R, const float* __restrict__ GZ,
              const float* __restrict__ Wdec, const float* __restrict__ bdec,
              const float* __restrict__ KL, const float* __restrict__ Z,
              float* __restrict__ pm, float* __restrict__ pk, unsigned tot) {
    float ms = 0.f, ks = 0.f;
    for (unsigned i = blockIdx.x * 256u + threadIdx.x; i < tot; i += gridDim.x * 256u) {
        unsigned l = i & 255u;
        unsigned row = i >> 8;
        float d = eluf(GZ[row] * Wdec[l] + bdec[l]);
        float diff = R[i] - d;
        ms += diff * diff;
        float z = Z[i];
        ks += KL[i] + 0.5f * z * z + 0.916290731874155f;   // + log(2.5)
    }
    __shared__ float sh[256];
    sh[threadIdx.x] = ms; __syncthreads();
    for (int o = 128; o > 0; o >>= 1) { if (threadIdx.x < o) sh[threadIdx.x] += sh[threadIdx.x + o]; __syncthreads(); }
    if (threadIdx.x == 0) pm[blockIdx.x] = sh[0];
    __syncthreads();
    sh[threadIdx.x] = ks; __syncthreads();
    for (int o = 128; o > 0; o >>= 1) { if (threadIdx.x < o) sh[threadIdx.x] += sh[threadIdx.x + o]; __syncthreads(); }
    if (threadIdx.x == 0) pk[blockIdx.x] = sh[0];
}

__global__ __launch_bounds__(256)
void k_final(const float* __restrict__ pm, const float* __restrict__ pk, int np,
             const float* __restrict__ NLL, const float* __restrict__ ACC,
             float* __restrict__ out, float invTot) {
    __shared__ float sh[256];
    int t = threadIdx.x;
    float sm = 0.f, sk = 0.f, sn = 0.f, sa = 0.f;
    for (int i = t; i < np; i += 256) { sm += pm[i]; sk += pk[i]; }
    if (t < 64) { sn = NLL[t]; sa = ACC[t]; }
    sh[t] = sm; __syncthreads();
    for (int o = 128; o > 0; o >>= 1) { if (t < o) sh[t] += sh[t + o]; __syncthreads(); }
    sm = sh[0]; __syncthreads();
    sh[t] = sk; __syncthreads();
    for (int o = 128; o > 0; o >>= 1) { if (t < o) sh[t] += sh[t + o]; __syncthreads(); }
    sk = sh[0]; __syncthreads();
    sh[t] = sn; __syncthreads();
    for (int o = 128; o > 0; o >>= 1) { if (t < o) sh[t] += sh[t + o]; __syncthreads(); }
    sn = sh[0]; __syncthreads();
    sh[t] = sa; __syncthreads();
    for (int o = 128; o > 0; o >>= 1) { if (t < o) sh[t] += sh[t + o]; __syncthreads(); }
    sa = sh[0];
    if (t == 0) {
        out[0] = sn / 64.f + sm * invTot + sk * invTot;
        out[1] = sa / 64.f;
    }
}

// ---------------------------------------------------------------------------
// Host-side launch sequence
// ---------------------------------------------------------------------------
extern "C" void kernel_launch(void* const* d_in, const int* in_sizes, int n_in,
                              void* d_out, int out_size, void* d_ws, size_t ws_size,
                              hipStream_t stream) {
    (void)in_sizes; (void)n_in; (void)out_size; (void)ws_size;

    const float* adj  = (const float*)d_in[0];
    const float* h    = (const float*)d_in[1];
    const float* eps  = (const float*)d_in[2];
    const int*   labels = (const int*)d_in[3];
    const float* Ws   = (const float*)d_in[4];
    const float* bs   = (const float*)d_in[5];
    const float* Wmu  = (const float*)d_in[6];
    const float* bmu  = (const float*)d_in[7];
    const float* Wlv  = (const float*)d_in[8];
    const float* blv  = (const float*)d_in[9];
    const float* Wvec = (const float*)d_in[10];
    const float* bvec = (const float*)d_in[11];
    const float* Wdec = (const float*)d_in[12];
    const float* bdec = (const float*)d_in[13];
    const float* W1   = (const float*)d_in[14];
    const float* b1   = (const float*)d_in[15];
    const float* W2   = (const float*)d_in[16];
    const float* b2   = (const float*)d_in[17];
    const float* beta = (const float*)d_in[18];
    const float* Wih  = (const float*)d_in[19];
    const float* Whh  = (const float*)d_in[20];
    const float* bih  = (const float*)d_in[21];
    const float* bhh  = (const float*)d_in[22];
    float* out = (float*)d_out;

    const int Bn = 64, Nn = 512, INd = 128, LD = 256;
    const size_t BN = (size_t)Bn * Nn;   // 32768 rows

    char* ws = (char*)d_ws;
    size_t cur = 0;
    auto alloc = [&](size_t bytes) -> char* {
        char* p = ws + cur;
        cur += (bytes + 255) & ~(size_t)255;
        return p;
    };

    unsigned short* adjBF  = (unsigned short*)alloc(BN * Nn * 2);      // 32 MB
    float*          invRS  = (float*)alloc(BN * 4);
    unsigned short* xT     = (unsigned short*)alloc(BN * LD * 2);      // 16 MB (reused)
    float*          big0   = (float*)alloc(BN * LD * 4);               // 32 MB (t0/t1/t2)
    unsigned short* big0BF = (unsigned short*)alloc(BN * LD * 2);
    float*          Rf     = (float*)alloc(BN * LD * 4);
    float*          MU     = (float*)alloc(BN * LD * 4);
    float*          LVKL   = (float*)alloc(BN * LD * 4);               // logvar -> kl
    float*          Zf     = (float*)alloc(BN * LD * 4);
    unsigned short* ZC     = (unsigned short*)alloc(BN * 512 * 2);     // [z | z0] bf16
    unsigned short* WC     = (unsigned short*)alloc((size_t)4 * 1024 * 512 * 2);
    unsigned short* WsT    = (unsigned short*)alloc((size_t)LD * INd * 2);
    unsigned short* WmuT   = (unsigned short*)alloc((size_t)LD * LD * 2);
    unsigned short* WlvT   = (unsigned short*)alloc((size_t)LD * LD * 2);
    float*          G      = (float*)alloc(BN * 1024 * 4);             // 128 MB
    float*          ZK     = (float*)alloc(BN * 4);
    float*          GZ     = (float*)alloc(BN * 4);
    float*          HF     = (float*)alloc((size_t)Bn * LD * 4);
    float*          NLLa   = (float*)alloc(64 * 4);
    float*          ACCa   = (float*)alloc(64 * 4);
    const int NB = 1024;
    float*          pm     = (float*)alloc(NB * 4);
    float*          pk     = (float*)alloc(NB * 4);

    // 1. adjacency -> bf16 (A operand, row-major) + inverse row sums
    { int tot = (int)(BN * Nn);
      k_f32_to_bf16<<<(tot + 255) / 256, 256, 0, stream>>>(adj, adjBF, tot); }
    k_rowsum_inv<<<(int)(BN / 8), 256, 0, stream>>>(adj, invRS);

    // 2. h^T bf16, then t0 = rowscale(adj @ h)       [B,512,128]
    { dim3 g(INd / 32, Nn / 32, Bn);
      k_transpose_bf16<<<g, dim3(32, 8), 0, stream>>>(h, xT, Nn, INd); }
    { dim3 g(INd / 128, Nn / 64, Bn);
      k_gemm_bf16<<<g, 256, 0, stream>>>(adjBF, (long long)Nn * Nn, Nn,
                                         xT, (long long)INd * Nn, Nn,
                                         big0, (long long)Nn * INd, INd,
                                         Nn, INd, Nn, invRS, Nn, nullptr, GF_ROWSCALE); }

    // 3. r = elu(t0 @ Ws + bs)                        [32768,256]
    { int tot = (int)(BN * INd);
      k_f32_to_bf16<<<(tot + 255) / 256, 256, 0, stream>>>(big0, big0BF, tot); }
    { dim3 g(LD / 32, INd / 32, 1);
      k_transpose_bf16<<<g, dim3(32, 8), 0, stream>>>(Ws, WsT, INd, LD); }
    { dim3 g(LD / 128, (int)(BN / 64), 1);
      k_gemm_bf16<<<g, 256, 0, stream>>>(big0BF, 0, INd, WsT, 0, INd, Rf, 0, LD,
                                         (int)BN, LD, INd, nullptr, 0, bs, GF_BIAS | GF_ELU); }

    // 4. t1 = rowscale(adj @ r); mu / logvar projections
    { dim3 g(LD / 32, Nn / 32, Bn);
      k_transpose_bf16<<<g, dim3(32, 8), 0, stream>>>(Rf, xT, Nn, LD); }
    { dim3 g(LD / 128, Nn / 64, Bn);
      k_gemm_bf16<<<g, 256, 0, stream>>>(adjBF, (long long)Nn * Nn, Nn,
                                         xT, (long long)LD * Nn, Nn,
                                         big0, (long long)Nn * LD, LD,
                                         Nn, LD, Nn, invRS, Nn, nullptr, GF_ROWSCALE); }
    { int tot = (int)(BN * LD);
      k_f32_to_bf16<<<(tot + 255) / 256, 256, 0, stream>>>(big0, big0BF, tot); }
    { dim3 g(LD / 32, LD / 32, 1);
      k_transpose_bf16<<<g, dim3(32, 8), 0, stream>>>(Wmu, WmuT, LD, LD);
      k_transpose_bf16<<<g, dim3(32, 8), 0, stream>>>(Wlv, WlvT, LD, LD); }
    { dim3 g(LD / 128, (int)(BN / 64), 1);
      k_gemm_bf16<<<g, 256, 0, stream>>>(big0BF, 0, LD, WmuT, 0, LD, MU, 0, LD,
                                         (int)BN, LD, LD, nullptr, 0, bmu, GF_BIAS | GF_ELU);
      k_gemm_bf16<<<g, 256, 0, stream>>>(big0BF, 0, LD, WlvT, 0, LD, LVKL, 0, LD,
                                         (int)BN, LD, LD, nullptr, 0, blv, GF_BIAS | GF_ELU); }

    // 5. reparameterize + KL init; pack flow weights; run 4 IAF steps
    { int tot = 4 * 1024 * 512;
      k_pack_wc<<<(tot + 255) / 256, 256, 0, stream>>>(Wih, Whh, WC, tot); }
    { int tot = (int)(BN * LD);
      k_z0_init<<<(tot + 255) / 256, 256, 0, stream>>>(MU, LVKL, eps, Zf, ZC, tot); }
    for (int t = 0; t < 4; ++t) {
        dim3 g(1024 / 128, (int)(BN / 64), 1);
        k_gemm_bf16<<<g, 256, 0, stream>>>(ZC, 0, 512,
                                           WC + (size_t)t * 1024 * 512, 0, 512,
                                           G, 0, 1024, (int)BN, 1024, 512,
                                           nullptr, 0, nullptr, 0);
        int tot = (int)(BN * LD);
        k_flow<<<(tot + 255) / 256, 256, 0, stream>>>(G, bih + t * 1024, bhh + t * 1024,
                                                      Zf, LVKL, ZC, tot);
    }

    // 6. t2 = rowscale(adj @ z); zv/zk; gz = rowscale(adj @ zk)
    { dim3 g(LD / 32, Nn / 32, Bn);
      k_transpose_bf16<<<g, dim3(32, 8), 0, stream>>>(Zf, xT, Nn, LD); }
    { dim3 g(LD / 128, Nn / 64, Bn);
      k_gemm_bf16<<<g, 256, 0, stream>>>(adjBF, (long long)Nn * Nn, Nn,
                                         xT, (long long)LD * Nn, Nn,
                                         big0, (long long)Nn * LD, LD,
                                         Nn, LD, Nn, invRS, Nn, nullptr, GF_ROWSCALE); }
    k_zv<<<(int)(BN / 8), 256, 0, stream>>>(big0, Wvec, bvec, beta, ZK, (int)BN);
    k_gz<<<(int)(BN / 8), 256, 0, stream>>>(adj, ZK, invRS, GZ);

    // 7. readout head + losses
    k_hfin<<<Bn, 256, 0, stream>>>(Rf, ZK, HF);
    k_head<<<Bn, 256, 0, stream>>>(HF, W1, b1, W2, b2, labels, NLLa, ACCa);
    k_mse_kl<<<NB, 256, 0, stream>>>(Rf, GZ, Wdec, bdec, LVKL, Zf, pm, pk,
                                     (unsigned)(BN * LD));
    k_final<<<1, 256, 0, stream>>>(pm, pk, NB, NLLa, ACCa, out,
                                   1.0f / (float)(BN * LD));
}
